// DifferentiableRiskBudgeting_41137196761697
// MI455X (gfx1250) — matrix operand order repaired and checked by reading, not compiled, not run on hardware
//
#include <hip/hip_runtime.h>
#include <stdint.h>

typedef _Float16 v8h  __attribute__((ext_vector_type(8)));
typedef _Float16 v16h __attribute__((ext_vector_type(16)));
typedef float    v8f  __attribute__((ext_vector_type(8)));

#define NB   512
#define P    256
#define MAXW 0.1f
#define EPSI 1e-8f
#define NPGD 250
#define NBIS 30
#define NPOW 20
#define SSTR 264   // padded row stride in halfs (row = 528 B -> 16B aligned, banks spread)

// LDS byte layout (all offsets 16B aligned)
#define OFF_SHI   0
#define OFF_SLO   (P * SSTR * 2)                  // 135168
#define OFF_W16HI (2 * P * SSTR * 2)              // 270336
#define OFF_W16LO (OFF_W16HI + 512)
#define OFF_Y     (OFF_W16LO + 512)
#define OFF_STAGE (OFF_Y + 1024)                  // 2 x 16KB double buffer
#define SMEM_BYTES (OFF_STAGE + 2 * 16384)        // 305152 B <= 320KB/WGP

__device__ __forceinline__ float wsum(float x) {
#pragma unroll
  for (int m = 16; m >= 1; m >>= 1) x += __shfl_xor(x, m, 32);
  return x;
}
__device__ __forceinline__ float wmin(float x) {
#pragma unroll
  for (int m = 16; m >= 1; m >>= 1) x = fminf(x, __shfl_xor(x, m, 32));
  return x;
}
__device__ __forceinline__ float wmax(float x) {
#pragma unroll
  for (int m = 16; m >= 1; m >>= 1) x = fmaxf(x, __shfl_xor(x, m, 32));
  return x;
}

__device__ __forceinline__ v16h cat16(v8h a, v8h b) {
  return __builtin_shufflevector(a, b, 0,1,2,3,4,5,6,7,8,9,10,11,12,13,14,15);
}

// y = S * w, S = Shi + Slo (f16 split) in LDS, w = whi + wlo (f16 split) in LDS.
// 8 waves x 2 row-tiles of 16 rows each. B operand = w chunk broadcast to all 16 cols,
// so every column of D equals y; lanes with (lane&15)==0 write y to LDS.
__device__ __forceinline__ void matvec(const _Float16* __restrict__ shi,
                                       const _Float16* __restrict__ slo,
                                       const _Float16* __restrict__ whi,
                                       const _Float16* __restrict__ wlo,
                                       float* __restrict__ yv,
                                       int wave, int lane)
{
  const int r  = lane & 15;
  const int hf = lane >> 4;
#pragma unroll
  for (int tt = 0; tt < 2; ++tt) {
    const int t = wave * 2 + tt;
    v8f acc = {0.f,0.f,0.f,0.f,0.f,0.f,0.f,0.f};
    const int rowoff = (t * 16 + r) * SSTR + 8 * hf;   // A: lanes0-15 K=c..c+7, lanes16-31 K=c+8..15
#pragma unroll
    for (int c = 0; c < 256; c += 32) {
      const _Float16* pah = shi + rowoff + c;
      const _Float16* pal = slo + rowoff + c;
      // elems0-7 = K chunk, elems8-15 = K chunk+16 (per 16-bit A-matrix 16x32 layout)
      v16h Ahi = cat16(*(const v8h*)pah, *(const v8h*)(pah + 16));
      v16h Alo = cat16(*(const v8h*)pal, *(const v8h*)(pal + 16));
      // B 32x16 broadcast: same K order as A: elems0-7 = K c+8hf..+7, elems8-15 = K c+16+8hf..+7 (LDS broadcast reads)
      const _Float16* pbh = whi + c + 8 * hf;
      const _Float16* pbl = wlo + c + 8 * hf;
      v16h Bhi = cat16(*(const v8h*)pbh, *(const v8h*)(pbh + 16));
      v16h Blo = cat16(*(const v8h*)pbl, *(const v8h*)(pbl + 16));
      acc = __builtin_amdgcn_wmma_f32_16x16x32_f16(false, Ahi, false, Bhi, (short)0, acc, false, false);
      acc = __builtin_amdgcn_wmma_f32_16x16x32_f16(false, Ahi, false, Blo, (short)0, acc, false, false);
      acc = __builtin_amdgcn_wmma_f32_16x16x32_f16(false, Alo, false, Bhi, (short)0, acc, false, false);
    }
    if (r == 0) {   // D: VGPR e = row e (lanes0-15) / row 8+e (lanes16-31)
#pragma unroll
      for (int e = 0; e < 8; ++e) yv[t * 16 + 8 * hf + e] = acc[e];
    }
  }
}

__device__ __forceinline__ void store_w16(const float* w8, _Float16* whi, _Float16* wlo, int lane) {
  v8h h, l;
#pragma unroll
  for (int j = 0; j < 8; ++j) {
    _Float16 hh = (_Float16)w8[j];
    h[j] = hh;
    l[j] = (_Float16)(w8[j] - (float)hh);
  }
  *(v8h*)(whi + 8 * lane) = h;
  *(v8h*)(wlo + 8 * lane) = l;
}

extern "C" __global__ __launch_bounds__(256)
void rb_solver(const float* __restrict__ sigma,
               const float* __restrict__ beta,
               const float* __restrict__ wprev,
               const float* __restrict__ loglam1,
               const float* __restrict__ loglam2,
               float* __restrict__ out)
{
  extern __shared__ char smem[];
  _Float16* shi = (_Float16*)(smem + OFF_SHI);
  _Float16* slo = (_Float16*)(smem + OFF_SLO);
  _Float16* whi = (_Float16*)(smem + OFF_W16HI);
  _Float16* wlo = (_Float16*)(smem + OFF_W16LO);
  float*    yv  = (float*)(smem + OFF_Y);

  const int b    = blockIdx.x;
  const int tid  = threadIdx.x;
  const int lane = tid & 31;
  const int wave = tid >> 5;

  const float lam1 = expf(loglam1[0]);
  const float lam2 = expf(loglam2[0]);

  // ---------------- stage sigma -> LDS (async, double buffered), convert to f16 hi/lo ----------------
  {
    const uint64_t gbase = (uint64_t)(uintptr_t)(sigma + (size_t)b * (P * P));
    const uint32_t stage_base = (uint32_t)(uintptr_t)(smem + OFF_STAGE);
#pragma unroll
    for (int i = 0; i < 4; ++i) {     // issue chunk 0 (16 rows = 16KB; 4 b128 per thread)
      uint32_t ldsa = stage_base + (uint32_t)(tid * 16 + i * 4096);
      uint32_t goff = (uint32_t)(tid * 16 + i * 4096);
      asm volatile("global_load_async_to_lds_b128 %0, %1, %2"
                   :: "v"(ldsa), "v"(goff), "s"(gbase) : "memory");
    }
    for (int ck = 0; ck < 16; ++ck) {
      if (ck + 1 < 16) {
#pragma unroll
        for (int i = 0; i < 4; ++i) { // prefetch next chunk into other buffer
          uint32_t ldsa = stage_base + (uint32_t)(((ck + 1) & 1) * 16384 + tid * 16 + i * 4096);
          uint32_t goff = (uint32_t)((ck + 1) * 16384 + tid * 16 + i * 4096);
          asm volatile("global_load_async_to_lds_b128 %0, %1, %2"
                       :: "v"(ldsa), "v"(goff), "s"(gbase) : "memory");
        }
        asm volatile("s_wait_asynccnt 0x4" ::: "memory");  // chunk ck complete (in-order)
      } else {
        asm volatile("s_wait_asynccnt 0x0" ::: "memory");
      }
      __syncthreads();
      { // convert 16 rows: this thread handles row ck*16 + tid/16, 16 consecutive cols
        const float* st = (const float*)(smem + OFF_STAGE + (ck & 1) * 16384);
        const int rloc = tid >> 4;
        const int c0   = (tid & 15) * 16;
        const int row  = ck * 16 + rloc;
        v8h h0, l0, h1, l1;
#pragma unroll
        for (int j = 0; j < 8; ++j) {
          float x = st[rloc * 256 + c0 + j];
          _Float16 hx = (_Float16)x;
          h0[j] = hx; l0[j] = (_Float16)(x - (float)hx);
          float y = st[rloc * 256 + c0 + 8 + j];
          _Float16 hy = (_Float16)y;
          h1[j] = hy; l1[j] = (_Float16)(y - (float)hy);
        }
        *(v8h*)(shi + row * SSTR + c0)     = h0;
        *(v8h*)(shi + row * SSTR + c0 + 8) = h1;
        *(v8h*)(slo + row * SSTR + c0)     = l0;
        *(v8h*)(slo + row * SSTR + c0 + 8) = l1;
      }
      __syncthreads();
    }
  }

  // ---------------- per-lane slices of beta / w_prev (8 contiguous elems per lane) ----------------
  float beta8[8], wp8[8];
  {
    const float* bp = beta  + (size_t)b * P + 8 * lane;
    const float* pp = wprev + (size_t)b * P + 8 * lane;
#pragma unroll
    for (int j = 0; j < 8; ++j) { beta8[j] = bp[j]; wp8[j] = pp[j]; }
  }

  // ---------------- power iteration: lambda_max estimate ----------------
  float v8[8];
#pragma unroll
  for (int j = 0; j < 8; ++j) v8[j] = 0.0625f;   // 1/sqrt(256)
  if (wave == 0) store_w16(v8, whi, wlo, lane);
  __syncthreads();

  for (int it = 0; it < NPOW; ++it) {
    matvec(shi, slo, whi, wlo, yv, wave, lane);
    __syncthreads();
    float u8[8], ss = 0.f;
#pragma unroll
    for (int j = 0; j < 8; ++j) { u8[j] = yv[8 * lane + j]; ss += u8[j] * u8[j]; }
    float nrm = sqrtf(wsum(ss)) + EPSI;          // every wave computes identically (redundant)
#pragma unroll
    for (int j = 0; j < 8; ++j) v8[j] = u8[j] / nrm;
    if (wave == 0) store_w16(v8, whi, wlo, lane);
    __syncthreads();
  }
  matvec(shi, slo, whi, wlo, yv, wave, lane);    // u = S v for lmax = v'Sv
  __syncthreads();
  float dd = 0.f;
#pragma unroll
  for (int j = 0; j < 8; ++j) dd += v8[j] * yv[8 * lane + j];
  const float lmax = wsum(dd);
  const float step = 1.0f / (2.0f * lmax + 2.0f * lam2 + 1e-6f);

  // ---------------- projected gradient descent ----------------
  float w8[8];
#pragma unroll
  for (int j = 0; j < 8; ++j) w8[j] = 1.0f / 256.0f;
  if (wave == 0) store_w16(w8, whi, wlo, lane);
  __syncthreads();

  for (int it = 0; it < NPGD; ++it) {
    matvec(shi, slo, whi, wlo, yv, wave, lane);
    __syncthreads();
    float vv8[8], vmn = 1e30f, vmx = -1e30f;
#pragma unroll
    for (int j = 0; j < 8; ++j) {
      float g = 2.0f * yv[8 * lane + j] - beta8[j] + lam1 + 2.0f * lam2 * (w8[j] - wp8[j]);
      float vv = w8[j] - step * g;
      vv8[j] = vv;
      vmn = fminf(vmn, vv); vmx = fmaxf(vmx, vv);
    }
    // capped-simplex projection by bisection; wave-local (redundant in all 8 waves, no barriers)
    float lo = wmin(vmn) - MAXW;
    float hi = wmax(vmx);
    for (int k = 0; k < NBIS; ++k) {
      float tau = 0.5f * (lo + hi);
      float ps = 0.f;
#pragma unroll
      for (int j = 0; j < 8; ++j) ps += fminf(fmaxf(vv8[j] - tau, 0.f), MAXW);
      float s = wsum(ps);
      if (s > 1.0f) lo = tau; else hi = tau;
    }
    float tau = 0.5f * (lo + hi);
#pragma unroll
    for (int j = 0; j < 8; ++j) w8[j] = fminf(fmaxf(vv8[j] - tau, 0.f), MAXW);
    if (wave == 0) store_w16(w8, whi, wlo, lane);
    __syncthreads();
  }

  // ---------------- clamp + renormalize, write out ----------------
  float ps = 0.f;
#pragma unroll
  for (int j = 0; j < 8; ++j) ps += w8[j];     // w already in [0, MAXW]
  float s = wsum(ps) + EPSI;
  if (wave == 0) {
    float* op = out + (size_t)b * P + 8 * lane;
#pragma unroll
    for (int j = 0; j < 8; ++j) op[j] = w8[j] / s;
  }
}

extern "C" void kernel_launch(void* const* d_in, const int* in_sizes, int n_in,
                              void* d_out, int out_size, void* d_ws, size_t ws_size,
                              hipStream_t stream) {
  (void)in_sizes; (void)n_in; (void)out_size; (void)d_ws; (void)ws_size;
  const float* sigma = (const float*)d_in[0];
  const float* beta  = (const float*)d_in[1];
  const float* wprev = (const float*)d_in[2];
  const float* ll1   = (const float*)d_in[3];
  const float* ll2   = (const float*)d_in[4];
  float* out = (float*)d_out;

  // allow > 64KB dynamic LDS (CDNA5 WGP has 320KB); deterministic & idempotent
  hipFuncSetAttribute((const void*)rb_solver,
                      hipFuncAttributeMaxDynamicSharedMemorySize, SMEM_BYTES);
  rb_solver<<<NB, 256, SMEM_BYTES, stream>>>(sigma, beta, wprev, ll1, ll2, out);
}
